// DSN_42941083025876
// MI455X (gfx1250) — matrix operand order repaired and checked by reading, not compiled, not run on hardware
//
#include <hip/hip_runtime.h>

typedef __attribute__((ext_vector_type(16))) _Float16   v16h;
typedef __attribute__((ext_vector_type(4)))  _Float16   v4h;
typedef __attribute__((ext_vector_type(8)))  float      v8f;
typedef __attribute__((ext_vector_type(4)))  unsigned   v4u;

namespace {

constexpr int BM  = 64;     // rows per workgroup tile (all launched M are multiples of 64)
constexpr int BN  = 64;     // cols per workgroup tile
constexpr int BK  = 32;     // K-step (matches 16x16x32 f16 WMMA)
constexpr int TPB = 128;    // 4 waves of 32
constexpr int LDK = BK + 8; // padded LDS pitch in halves (80B rows keep 16B-aligned chunks)
constexpr int VPT = (BM * BK / 4) / TPB;   // float4 fetches per thread per f32 tile = 4
constexpr int APT = (BM * BK / 8) / TPB;   // 16B async chunks per thread per f16 tile = 2
constexpr int NW  = 10;     // number of WMMA weight matrices

union Frag16 { v16h h; v4u u[2]; };

__device__ __forceinline__ void wait_async0() {
    asm volatile("s_wait_asynccnt 0x0" ::: "memory");
}

// DMA one 16B chunk global->LDS (ASYNCcnt-tracked).
__device__ __forceinline__ void async_ld_b128(unsigned lds_off, const void* gaddr) {
    asm volatile("global_load_async_to_lds_b128 %0, %1, off"
                 :: "v"(lds_off), "v"(gaddr) : "memory");
}

// Stage a 64x32 f16 tile via async DMA with K-tail guard (K % 8 == 0 for all
// f16 activations, so each 8-half chunk is fully valid or fully OOB -> zeroed).
__device__ __forceinline__ void async_stage_guard(const _Float16* __restrict__ A,
                                                  int r0, int k0, int K, int tid,
                                                  _Float16 (* __restrict__ dst)[LDK])
{
    #pragma unroll
    for (int i = 0; i < APT; ++i) {
        int cid = tid + TPB * i;       // 0..255
        int row = cid >> 2;            // 4 chunks per row
        int c8  = (cid & 3) << 3;
        int gk  = k0 + c8;
        unsigned lds = (unsigned)(size_t)&dst[row][c8];
        if (gk < K) {
            async_ld_b128(lds, (const void*)(A + (size_t)(r0 + row) * K + gk));
        } else {
            v4u z = {0u, 0u, 0u, 0u};
            *(v4u*)&dst[row][c8] = z;
        }
    }
}

// Guard-free async staging (pre-padded f16 weights: rows %64 == 0, Kp %32 == 0).
__device__ __forceinline__ void async_stage_full(const _Float16* __restrict__ W,
                                                 int r0, int k0, int Kp, int tid,
                                                 _Float16 (* __restrict__ dst)[LDK])
{
    #pragma unroll
    for (int i = 0; i < APT; ++i) {
        int cid = tid + TPB * i;
        int row = cid >> 2;
        int c8  = (cid & 3) << 3;
        unsigned lds = (unsigned)(size_t)&dst[row][c8];
        async_ld_b128(lds, (const void*)(W + (size_t)(r0 + row) * Kp + k0 + c8));
    }
}

// Fetch one 64x32 f32 tile as float4s into registers (K % 4 == 0 -> aligned,
// all-or-nothing OOB; rows always valid since M % 64 == 0).
__device__ __forceinline__ void fetch_tile(const float* __restrict__ P, int r0,
                                           int k0, int K, int tid, float4* reg)
{
    #pragma unroll
    for (int i = 0; i < VPT; ++i) {
        int vid = tid + TPB * i;        // 0..511
        int row = vid >> 3;             // 8 float4 per row
        int c4  = (vid & 7) << 2;
        int gk  = k0 + c4;
        float4 v = make_float4(0.f, 0.f, 0.f, 0.f);
        if (gk < K) v = *(const float4*)(P + (long)(r0 + row) * K + gk);
        reg[i] = v;
    }
}

__device__ __forceinline__ void store_tile(_Float16 (* __restrict__ dst)[LDK],
                                           int tid, const float4* reg)
{
    #pragma unroll
    for (int i = 0; i < VPT; ++i) {
        int vid = tid + TPB * i;
        int row = vid >> 3;
        int c4  = (vid & 7) << 2;
        v4h h;
        h.x = (_Float16)reg[i].x; h.y = (_Float16)reg[i].y;
        h.z = (_Float16)reg[i].z; h.w = (_Float16)reg[i].w;
        *(v4h*)&dst[row][c4] = h;   // ds_store_b64, 8B aligned
    }
}

// C[M,Nn] = act(A[M,K] @ W[Nn,K]^T + bias); requires M % 64 == 0 (true for all uses).
// TA: float (cvt-at-staging) or _Float16 (async DMA staging).
// Wp: pre-converted f16 weights, zero-padded to [Np x Kp], Np%64==0, Kp%32==0.
template<typename TA, typename TC, bool RELU>
__global__ __launch_bounds__(TPB)
void wmma_gemm_bias(const TA* __restrict__ A, const _Float16* __restrict__ Wp,
                    const float* __restrict__ bias, TC* __restrict__ C,
                    int M, int Nn, int K, int Kp)
{
    constexpr bool AF16 = (sizeof(TA) == 2);

    __shared__ __align__(16) _Float16 As[2][BM][LDK];
    __shared__ __align__(16) _Float16 Ws[2][BN][LDK];

    const int m0   = blockIdx.x * BM;
    const int n0   = blockIdx.y * BN;
    const int tid  = threadIdx.x;
    const int wave = tid >> 5;
    const int lane = tid & 31;
    const int g    = lane >> 4;  // K-group selector within fragment
    const int hl   = lane & 15;  // row (A) / col (B,C) within 16

    v8f acc[4] = {};

    const int ksteps = Kp / BK;
    float4 ra[VPT];

    // Prologue: stage tile 0
    if constexpr (AF16) {
        async_stage_guard((const _Float16*)A, m0, 0, K, tid, As[0]);
    } else {
        fetch_tile((const float*)A, m0, 0, K, tid, ra);
        store_tile(As[0], tid, ra);
    }
    async_stage_full(Wp, n0, 0, Kp, tid, Ws[0]);
    wait_async0();
    __syncthreads();

    for (int ks = 0; ks < ksteps; ++ks) {
        const int  p    = ks & 1;
        const bool more = (ks + 1) < ksteps;

        // Prefetch next tile (overlaps with WMMA below)
        if (more) {
            const int kn = (ks + 1) * BK;
            if constexpr (AF16) {
                async_stage_guard((const _Float16*)A, m0, kn, K, tid, As[1 - p]);
            } else {
                fetch_tile((const float*)A, m0, kn, K, tid, ra);
            }
            async_stage_full(Wp, n0, kn, Kp, tid, Ws[1 - p]);
        }

        // A fragment per CDNA5 16-bit A layout:
        //   lane hl = row M; halves e<8 -> K = 8g+e ; e>=8 -> K = 16+8g+(e-8)
        Frag16 a;
        {
            const _Float16* arow = &As[p][wave * 16 + hl][0];
            a.u[0] = *(const v4u*)(arow + 8 * g);
            a.u[1] = *(const v4u*)(arow + 16 + 8 * g);
        }
        // All 4 B fragments, then the 4 WMMAs
        Frag16 b[4];
        #pragma unroll
        for (int t = 0; t < 4; ++t) {
            const _Float16* brow = &Ws[p][16 * t + hl][8 * g];
            b[t].u[0] = *(const v4u*)(brow);
            b[t].u[1] = *(const v4u*)(brow + 16);
        }
        #pragma unroll
        for (int t = 0; t < 4; ++t) {
            acc[t] = __builtin_amdgcn_wmma_f32_16x16x32_f16(
                false, a.h, false, b[t].h, (short)0, acc[t], false, false);
        }

        if (more) {
            if constexpr (!AF16) store_tile(As[1 - p], tid, ra);
            wait_async0();
            __syncthreads();
        }
    }

    // Epilogue: C layout -> lane hl = col, VGPR r: row = 8g + r. No row guard (M%64==0).
    const long rowbase = m0 + wave * 16 + 8 * g;
    #pragma unroll
    for (int t = 0; t < 4; ++t) {
        const int col = n0 + 16 * t + hl;
        if (col < Nn) {
            const float bv = bias[col];
            #pragma unroll
            for (int r = 0; r < 8; ++r) {
                float v = acc[t][r] + bv;
                if (RELU) v = v > 0.0f ? v : 0.0f;
                C[(rowbase + r) * Nn + col] = (TC)v;
            }
        }
    }
}

// ---- one-shot weight conversion: f32 [Nn x K] -> f16 zero-padded [Np x Kp] ----
struct WSeg { const float* src; long base; int Nn, K, Kp; };
struct WPack { WSeg s[NW]; long total; _Float16* dst; };

__global__ void cvt_weights_kernel(WPack P)
{
    long i = (long)blockIdx.x * blockDim.x + threadIdx.x;
    if (i >= P.total) return;
    int si = 0;
    #pragma unroll
    for (int j = 1; j < NW; ++j)
        if (i >= P.s[j].base) si = j;
    const WSeg sg = P.s[si];
    long off = i - sg.base;
    int n = (int)(off / sg.Kp);
    int k = (int)(off % sg.Kp);
    _Float16 v = (_Float16)0.f;
    if (n < sg.Nn && k < sg.K) v = (_Float16)sg.src[(long)n * sg.K + k];
    P.dst[i] = v;
}

// out = mu + eps * exp(var/2), vectorized x4 (n % 4 == 0)
__global__ void reparam_kernel(const float* __restrict__ mu,
                               const float* __restrict__ var,
                               const float* __restrict__ eps,
                               float* __restrict__ out, long n4)
{
    long i = (long)blockIdx.x * blockDim.x + threadIdx.x;
    if (i >= n4) return;
    float4 m = ((const float4*)mu)[i];
    float4 v = ((const float4*)var)[i];
    float4 e = ((const float4*)eps)[i];
    float4 o;
    o.x = m.x + e.x * __expf(v.x * 0.5f);
    o.y = m.y + e.y * __expf(v.y * 0.5f);
    o.z = m.z + e.z * __expf(v.z * 0.5f);
    o.w = m.w + e.w * __expf(v.w * 0.5f);
    ((float4*)out)[i] = o;
}

// FIELD_SCALES: per-100 chunks cycling [0.3, 0.5, 0.1, 0.1]; 100 % 4 == 0 so a
// float4 never straddles a chunk boundary. Writes f16 for the wc1 GEMM.
__global__ void scale_kernel(const float* __restrict__ x,
                             _Float16* __restrict__ out, long n4)
{
    long i = (long)blockIdx.x * blockDim.x + threadIdx.x;
    if (i >= n4) return;
    const float tbl[4] = {0.3f, 0.5f, 0.1f, 0.1f};
    float s = tbl[((i * 4) / 100) & 3];
    float4 v = ((const float4*)x)[i];
    v4h h;
    h.x = (_Float16)(v.x * s); h.y = (_Float16)(v.y * s);
    h.z = (_Float16)(v.z * s); h.w = (_Float16)(v.w * s);
    ((v4h*)out)[i] = h;
}

// 2-output linear head, optional fused log_softmax (axis of size 2). K % 4 == 0.
template<typename TH, bool LOGSOFTMAX>
__global__ void head2_kernel(const TH* __restrict__ H,
                             const float* __restrict__ W2,
                             const float* __restrict__ b2,
                             float* __restrict__ out, int M, int K)
{
    int row = blockIdx.x * blockDim.x + threadIdx.x;
    if (row >= M) return;
    const float4* w0 = (const float4*)W2;
    const float4* w1 = (const float4*)(W2 + K);
    float a0 = b2[0], a1 = b2[1];
    const int k4 = K >> 2;
    for (int k = 0; k < k4; ++k) {
        float x0, x1, x2, x3;
        if constexpr (sizeof(TH) == 2) {
            v4h x = ((const v4h*)(H + (long)row * K))[k];
            x0 = (float)x.x; x1 = (float)x.y; x2 = (float)x.z; x3 = (float)x.w;
        } else {
            float4 x = ((const float4*)(H + (long)row * K))[k];
            x0 = x.x; x1 = x.y; x2 = x.z; x3 = x.w;
        }
        float4 u = w0[k], v = w1[k];
        a0 += x0 * u.x + x1 * u.y + x2 * u.z + x3 * u.w;
        a1 += x0 * v.x + x1 * v.y + x2 * v.z + x3 * v.w;
    }
    if (LOGSOFTMAX) {
        float m   = fmaxf(a0, a1);
        float lse = m + __logf(__expf(a0 - m) + __expf(a1 - m));
        a0 -= lse; a1 -= lse;
    }
    out[(long)row * 2 + 0] = a0;
    out[(long)row * 2 + 1] = a1;
}

inline long cdivl(long a, long b) { return (a + b - 1) / b; }
inline int  pad64(int x) { return (int)((x + 63) / 64) * 64; }
inline int  pad32(int x) { return (int)((x + 31) / 32) * 32; }

} // namespace

extern "C" void kernel_launch(void* const* d_in, const int* in_sizes, int n_in,
                              void* d_out, int out_size, void* d_ws, size_t ws_size,
                              hipStream_t stream)
{
    (void)n_in; (void)out_size; (void)ws_size;

    const float* input = (const float*)d_in[0];
    const float* eps1  = (const float*)d_in[1];
    const float* eps2  = (const float*)d_in[2];
    // d_in[3] = field_size (device scalar) -- derived from sizes instead
    const float* wsrc[NW] = {
        (const float*)d_in[4],   // ws_w   200x300
        (const float*)d_in[6],   // fc1_w  100x200
        (const float*)d_in[8],   // fc2_w  100x200
        (const float*)d_in[10],  // wsh_w  200x300
        (const float*)d_in[12],  // fc5_w  100x200
        (const float*)d_in[14],  // fc6_w  100x200
        (const float*)d_in[16],  // wd1_w  100x100
        (const float*)d_in[20],  // wc1_w  300x(field*100)
        (const float*)d_in[24],  // dec1_w 200x100
        (const float*)d_in[26],  // dec2_w 300x200
    };
    const float* ws_b   = (const float*)d_in[5];
    const float* fc1_b  = (const float*)d_in[7];
    const float* fc2_b  = (const float*)d_in[9];
    const float* wsh_b  = (const float*)d_in[11];
    const float* fc5_b  = (const float*)d_in[13];
    const float* fc6_b  = (const float*)d_in[15];
    const float* wd1_b  = (const float*)d_in[17];
    const float* wd2_w  = (const float*)d_in[18];
    const float* wd2_b  = (const float*)d_in[19];
    const float* wc1_b  = (const float*)d_in[21];
    const float* wc2_w  = (const float*)d_in[22];
    const float* wc2_b  = (const float*)d_in[23];
    const float* dec1_b = (const float*)d_in[25];
    const float* dec2_w = (const float*)d_in[27]; (void)dec2_w;
    const float* dec2_b = (const float*)d_in[27];

    const long N  = (long)in_sizes[0] / 300;   // 131072
    const int  Kc = in_sizes[20] / 300;        // field*100 = 800
    const long Mc = (N * 100) / Kc;            // N/field = 16384

    float* out      = (float*)d_out;
    float* o_priv   = out;
    float* o_shared = o_priv   + N * 100;
    float* o_dom    = o_shared + N * 100;
    float* o_class  = o_dom    + N * 2;
    float* o_rec    = o_class  + Mc * 2;
    float* o_mu1    = o_rec    + N * 300;
    float* o_var1   = o_mu1    + N * 100;
    float* o_mu2    = o_var1   + N * 100;
    float* o_var2   = o_mu2    + N * 100;

    _Float16* ws0h = (_Float16*)d_ws;       // activations: up to 200*N halves (52 MB)
    _Float16* ws1h = ws0h + N * 100;        // class hidden (300*Mc halves) overlay
    _Float16* wpad = ws0h + N * 200;        // padded f16 weights (~1.3 MB)

    // ---- build weight-conversion pack & padded pointers ----
    const int wNn[NW] = {200, 100, 100, 200, 100, 100, 100, 300, 200, 300};
    const int wK [NW] = {300, 200, 200, 300, 200, 200, 100, Kc,  100, 200};
    WPack pack;
    _Float16* wp[NW];
    int wKp[NW];
    long base = 0;
    for (int j = 0; j < NW; ++j) {
        int Np = pad64(wNn[j]);
        int Kp = pad32(wK[j]);
        pack.s[j].src = wsrc[j];
        pack.s[j].base = base;
        pack.s[j].Nn = wNn[j];
        pack.s[j].K  = wK[j];
        pack.s[j].Kp = Kp;
        wp[j]  = wpad + base;
        wKp[j] = Kp;
        base  += (long)Np * Kp;
    }
    pack.total = base;
    pack.dst   = wpad;

    const dim3 blk(TPB);
    auto ggrid = [](long M, long Nn) {
        return dim3((unsigned)cdivl(M, BM), (unsigned)cdivl(Nn, BN));
    };
    const int ETPB = 256;
    const long n4 = (N * 100) / 4;
    const dim3 egrid((unsigned)cdivl(n4, ETPB));

    // ---- weight conversion (tiny, once, padded & zero-filled) ----
    cvt_weights_kernel<<<(unsigned)cdivl(pack.total, ETPB), ETPB, 0, stream>>>(pack);

    // ---- private branch ----
    wmma_gemm_bias<float, _Float16, true ><<<ggrid(N, 200), blk, 0, stream>>>(input, wp[0], ws_b,  ws0h,   (int)N, 200, 300, wKp[0]);
    wmma_gemm_bias<_Float16, float, false><<<ggrid(N, 100), blk, 0, stream>>>(ws0h,  wp[1], fc1_b, o_mu1,  (int)N, 100, 200, wKp[1]);
    wmma_gemm_bias<_Float16, float, false><<<ggrid(N, 100), blk, 0, stream>>>(ws0h,  wp[2], fc2_b, o_var1, (int)N, 100, 200, wKp[2]);
    reparam_kernel<<<egrid, ETPB, 0, stream>>>(o_mu1, o_var1, eps1, o_priv, n4);

    // ---- shared branch ----
    wmma_gemm_bias<float, _Float16, true ><<<ggrid(N, 200), blk, 0, stream>>>(input, wp[3], wsh_b, ws0h,   (int)N, 200, 300, wKp[3]);
    wmma_gemm_bias<_Float16, float, false><<<ggrid(N, 100), blk, 0, stream>>>(ws0h,  wp[4], fc5_b, o_mu2,  (int)N, 100, 200, wKp[4]);
    wmma_gemm_bias<_Float16, float, false><<<ggrid(N, 100), blk, 0, stream>>>(ws0h,  wp[5], fc6_b, o_var2, (int)N, 100, 200, wKp[5]);
    reparam_kernel<<<egrid, ETPB, 0, stream>>>(o_mu2, o_var2, eps2, o_shared, n4);

    // ---- domain head (grad-reverse is identity in forward) ----
    wmma_gemm_bias<float, _Float16, true ><<<ggrid(N, 100), blk, 0, stream>>>(o_shared, wp[6], wd1_b, ws0h, (int)N, 100, 100, wKp[6]);
    head2_kernel<_Float16, false><<<(unsigned)cdivl(N, ETPB), ETPB, 0, stream>>>(ws0h, wd2_w, wd2_b, o_dom, (int)N, 100);

    // ---- class head: scale -> wc1 -> relu -> wc2 -> log_softmax ----
    scale_kernel<<<egrid, ETPB, 0, stream>>>(o_shared, ws0h, n4);
    wmma_gemm_bias<_Float16, _Float16, true ><<<ggrid(Mc, 300), blk, 0, stream>>>(ws0h, wp[7], wc1_b, ws1h, (int)Mc, 300, Kc, wKp[7]);
    head2_kernel<_Float16, true><<<(unsigned)cdivl(Mc, ETPB), ETPB, 0, stream>>>(ws1h, wc2_w, wc2_b, o_class, (int)Mc, 300);

    // ---- reconstruction ----
    wmma_gemm_bias<float, _Float16, true ><<<ggrid(N, 200), blk, 0, stream>>>(o_shared, wp[8], dec1_b, ws0h,  (int)N, 200, 100, wKp[8]);
    wmma_gemm_bias<_Float16, float, false><<<ggrid(N, 300), blk, 0, stream>>>(ws0h,     wp[9], dec2_b, o_rec, (int)N, 300, 200, wKp[9]);
}
